// ExperimentalEncoder_1967095021764
// MI455X (gfx1250) — matrix operand order repaired and checked by reading, not compiled, not run on hardware
//
#include <hip/hip_runtime.h>
#include <math.h>

#define Bk   8
#define Sk   32
#define Nk   1024
#define Fk   4
#define Hk   128
#define Ck   132        // F+H
#define CPk  160        // padded F+H (multiple of 32)
#define G3k  384        // 3H
#define GOk  512        // combined gemm output: 3H (W1) + H (W2), also 4H decoder gates
#define BNk  (Bk*Nk)    // 8192
#define NHk  (Nk*Hk)    // 131072

typedef __bf16 bfloat;
typedef __bf16 v16bf __attribute__((ext_vector_type(16)));
typedef __bf16 v8bf  __attribute__((ext_vector_type(8)));
typedef float  v8f   __attribute__((ext_vector_type(8)));

// ---------------------------------------------------------------- helpers

__device__ __forceinline__ float sigmf(float x) { return 1.0f / (1.0f + __expf(-x)); }

__device__ __forceinline__ v8f wmma_bf16(v16bf a, v16bf b, v8f c) {
  // D(16x16,f32) = A(16x32,bf16) * B(32x16,bf16) + C
  return __builtin_amdgcn_wmma_f32_16x16x32_bf16(false, a, false, b, (short)0, c,
                                                 false, false);
}

// A-fragment (16x32 bf16, row-major global, leading dim ld):
// lanes 0-15 hold row=lane, K base 0; lanes 16-31 hold row=lane-16, K base 8.
// vector elems 0..7 -> K=base..base+7, elems 8..15 -> K=base+16..base+23.
__device__ __forceinline__ v16bf load_a_frag(const bfloat* __restrict__ p0, int ld) {
  const int lane = threadIdx.x & 31;
  const int r    = lane & 15;
  const int koff = (lane >> 4) << 3;
  const bfloat* p = p0 + (size_t)r * ld + koff;
  v8bf lo = *(const v8bf*)(p);
  v8bf hi = *(const v8bf*)(p + 16);
  v16bf v;
#pragma unroll
  for (int i = 0; i < 8; ++i) { v[i] = lo[i]; v[8 + i] = hi[i]; }
  return v;
}

// B-fragment (32x16 bf16) from an LDS tile stored transposed:
// bT[c][k], k contiguous, row stride 40 elements (80B, 16B-aligned, spreads banks).
// lanes 0-15: col=lane, K=0..7,16..23 ; lanes 16-31: col=lane-16, K=8..15,24..31.
__device__ __forceinline__ v16bf load_b_frag(const bfloat* bT) {
  const int lane = threadIdx.x & 31;
  const int c    = lane & 15;
  const int kb   = (lane >> 4) << 3;
  const bfloat* p = bT + c * 40 + kb;
  v8bf lo = *(const v8bf*)(p);
  v8bf hi = *(const v8bf*)(p + 16);
  v16bf v;
#pragma unroll
  for (int i = 0; i < 8; ++i) { v[i] = lo[i]; v[8 + i] = hi[i]; }
  return v;
}

// ---------------------------------------------------------------- prep kernels

__global__ void k_cvt_f2b(const float* __restrict__ in, bfloat* __restrict__ out, int n) {
  int i = blockIdx.x * blockDim.x + threadIdx.x;
  if (i < n) out[i] = (bfloat)in[i];
}
__global__ void k_zero_f32(float* p, int n) {
  int i = blockIdx.x * blockDim.x + threadIdx.x;
  if (i < n) p[i] = 0.0f;
}
__global__ void k_zero_b16(bfloat* p, int n) {
  int i = blockIdx.x * blockDim.x + threadIdx.x;
  if (i < n) p[i] = (bfloat)0.0f;
}
// Wcat (CPk x GOk) = [W1 (132x384) | W2 (132x128)], rows 132..159 zero-padded.
__global__ void k_prep_wcat(const float* __restrict__ W1, const float* __restrict__ W2,
                            bfloat* __restrict__ wcat) {
  int idx = blockIdx.x * blockDim.x + threadIdx.x;
  if (idx >= CPk * GOk) return;
  int k = idx / GOk, c = idx - k * GOk;
  float v = 0.0f;
  if (k < Ck) v = (c < G3k) ? W1[k * G3k + c] : W2[k * Hk + (c - G3k)];
  wcat[idx] = (bfloat)v;
}
__global__ void k_prep_bias(const float* __restrict__ b1, const float* __restrict__ b2,
                            const float* __restrict__ bih, const float* __restrict__ bhh,
                            float* __restrict__ bias_cat, float* __restrict__ bias_dec) {
  int i = blockIdx.x * blockDim.x + threadIdx.x;
  if (i < GOk) {
    bias_cat[i] = (i < G3k) ? b1[i] : b2[i - G3k];
    bias_dec[i] = bih[i] + bhh[i];
  }
}

// ---------------------------------------------------------------- encoder kernels

// cat (B,N,CPk) bf16 = [ x[:,t,:,:] | hid | 0-pad ]
__global__ __launch_bounds__(256) void k_build_cat(const float* __restrict__ x,
                                                   const float* __restrict__ hid,
                                                   bfloat* __restrict__ cat, int tstep) {
  int idx = blockIdx.x * blockDim.x + threadIdx.x;  // < B*N*CPk
  int b = idx / (Nk * CPk);
  int r = idx - b * (Nk * CPk);
  int n = r / CPk;
  int c = r - n * CPk;
  float v;
  if (c < Fk)      v = x[(((size_t)b * Sk + tstep) * Nk + n) * Fk + c];
  else if (c < Ck) v = hid[(size_t)b * NHk + n * Hk + (c - Fk)];
  else             v = 0.0f;
  cat[idx] = (bfloat)v;
}

// Stage a 32(K) x 32(col) tile of a row-major matrix into LDS transposed:
// sB[c][k], vectorized b128 global reads (threads 0..127 active).
__device__ __forceinline__ void stage_bT(const bfloat* __restrict__ src, int ld,
                                         int k0, int c0, bfloat* sB, int t) {
  if (t < 128) {
    int k   = t >> 2;        // 0..31
    int cs8 = (t & 3) * 8;   // 0,8,16,24
    v8bf row = *(const v8bf*)(src + (size_t)(k0 + k) * ld + c0 + cs8);
#pragma unroll
    for (int i = 0; i < 8; ++i) sB[(cs8 + i) * 40 + k] = row[i];
  }
}

// AC = adj @ cat per batch: (1024x1024)x(1024x160), K=1024. Output bf16.
// Block: 256 thr (8 waves), tile 256 rows x 32 cols; wave w -> rows w*32..+31
// as two A-fragments sharing the two LDS B-fragments (4 WMMAs / K-chunk).
__global__ __launch_bounds__(256) void k_gemm_adj(const bfloat* __restrict__ adjb,
                                                  const bfloat* __restrict__ cat,
                                                  bfloat* __restrict__ ac) {
  const int bb   = blockIdx.z;
  const int m0   = blockIdx.y * 256;
  const int cb0  = blockIdx.x * 32;
  const int t    = threadIdx.x;
  const int wave = t >> 5;
  const int lane = t & 31;
  __shared__ alignas(16) bfloat sB[32 * 40];
  const bfloat* catB = cat + (size_t)bb * (Nk * CPk);
  const int mrow = m0 + wave * 32;
  v8f acc00 = {0.f,0.f,0.f,0.f,0.f,0.f,0.f,0.f};
  v8f acc01 = acc00, acc10 = acc00, acc11 = acc00;
  for (int k0 = 0; k0 < Nk; k0 += 32) {
    stage_bT(catB, CPk, k0, cb0, sB, t);
    __syncthreads();
    if (k0 + 32 < Nk) {
      __builtin_prefetch(adjb + (size_t)mrow * Nk + k0 + 32, 0, 1);
      __builtin_prefetch(adjb + (size_t)(mrow + 16) * Nk + k0 + 32, 0, 1);
    }
    v16bf a0 = load_a_frag(adjb + (size_t)mrow * Nk + k0, Nk);
    v16bf a1 = load_a_frag(adjb + (size_t)(mrow + 16) * Nk + k0, Nk);
    v16bf b0 = load_b_frag(sB);
    v16bf b1 = load_b_frag(sB + 16 * 40);
    acc00 = wmma_bf16(a0, b0, acc00);
    acc01 = wmma_bf16(a0, b1, acc01);
    acc10 = wmma_bf16(a1, b0, acc10);
    acc11 = wmma_bf16(a1, b1, acc11);
    __syncthreads();
  }
  bfloat* outp = ac + (size_t)bb * (Nk * CPk);
  const int cc = lane & 15, hi = lane >> 4;
#pragma unroll
  for (int j = 0; j < 8; ++j) {  // C layout: VGPR j -> row j (+8 for hi half), col = lane&15
    int r0 = mrow + j + 8 * hi;
    int r1 = r0 + 16;
    outp[(size_t)r0 * CPk + cb0 + cc]      = (bfloat)acc00[j];
    outp[(size_t)r0 * CPk + cb0 + 16 + cc] = (bfloat)acc01[j];
    outp[(size_t)r1 * CPk + cb0 + cc]      = (bfloat)acc10[j];
    outp[(size_t)r1 * CPk + cb0 + 16 + cc] = (bfloat)acc11[j];
  }
}

// G = AC(8192x160) @ Wcat(160x512) + bias; K=160 (5 chunks).
__global__ __launch_bounds__(256) void k_gemm_w(const bfloat* __restrict__ acf,
                                                const bfloat* __restrict__ wcat,
                                                const float* __restrict__ bias,
                                                float* __restrict__ G) {
  const int m0   = blockIdx.y * 256;
  const int cb0  = blockIdx.x * 32;
  const int t    = threadIdx.x;
  const int wave = t >> 5;
  const int lane = t & 31;
  __shared__ alignas(16) bfloat sB[32 * 40];
  const int mrow = m0 + wave * 32;
  v8f acc00 = {0.f,0.f,0.f,0.f,0.f,0.f,0.f,0.f};
  v8f acc01 = acc00, acc10 = acc00, acc11 = acc00;
  for (int k0 = 0; k0 < CPk; k0 += 32) {
    stage_bT(wcat, GOk, k0, cb0, sB, t);
    __syncthreads();
    v16bf a0 = load_a_frag(acf + (size_t)mrow * CPk + k0, CPk);
    v16bf a1 = load_a_frag(acf + (size_t)(mrow + 16) * CPk + k0, CPk);
    v16bf b0 = load_b_frag(sB);
    v16bf b1 = load_b_frag(sB + 16 * 40);
    acc00 = wmma_bf16(a0, b0, acc00);
    acc01 = wmma_bf16(a0, b1, acc01);
    acc10 = wmma_bf16(a1, b0, acc10);
    acc11 = wmma_bf16(a1, b1, acc11);
    __syncthreads();
  }
  const int cc = lane & 15, hi = lane >> 4;
#pragma unroll
  for (int j = 0; j < 8; ++j) {
    int r0 = mrow + j + 8 * hi;
    int r1 = r0 + 16;
    int c0 = cb0 + cc, c1 = cb0 + 16 + cc;
    G[(size_t)r0 * GOk + c0] = acc00[j] + bias[c0];
    G[(size_t)r0 * GOk + c1] = acc01[j] + bias[c1];
    G[(size_t)r1 * GOk + c0] = acc10[j] + bias[c0];
    G[(size_t)r1 * GOk + c1] = acc11[j] + bias[c1];
  }
}

// Encoder gating with the flat-split remap; cell state is identically zero (torch
// bug preserved), so hnew = sig(og_flat) * tanh(sig(ig_flat) * tanh(cs)).
// Also accumulates ctx += hnew (softmax over size-1 axis == 1 -> ctx = sum_t h_t).
__global__ __launch_bounds__(256) void k_gates(const float* __restrict__ G,
                                               float* __restrict__ hid,
                                               float* __restrict__ ctx) {
  const int b = blockIdx.y;
  const int p = blockIdx.x * blockDim.x + threadIdx.x;  // < NHk
  const float* Gb = G + (size_t)b * Nk * GOk;
  int j1 = NHk + p;       // ig slice of flattened (N,3H)
  int j2 = 2 * NHk + p;   // og slice
  float igv = sigmf(Gb[(size_t)(j1 / G3k) * GOk + (j1 % G3k)]);
  float ogv = sigmf(Gb[(size_t)(j2 / G3k) * GOk + (j2 % G3k)]);
  float cs  = tanhf(Gb[(size_t)(p >> 7) * GOk + G3k + (p & (Hk - 1))]);
  float h = ogv * tanhf(igv * cs);
  size_t o = (size_t)b * NHk + p;
  hid[o] = h;
  ctx[o] += h;
}

// ---------------------------------------------------------------- decoder kernels

// out = A(8192x128) @ W^T (W is 512x128 row-major) [+ bias] [+ addrow]; K=128.
// B[k][c] = W[c][k]: the "transposed" LDS tile is a straight row copy of W,
// so staging is a pure b128 -> b128 copy.
__global__ __launch_bounds__(256) void k_gemm_dec(const bfloat* __restrict__ A,
                                                  const bfloat* __restrict__ W,
                                                  float* __restrict__ out,
                                                  const float* __restrict__ addrow,
                                                  const float* __restrict__ bias) {
  const int m0   = blockIdx.y * 256;
  const int cb0  = blockIdx.x * 32;
  const int t    = threadIdx.x;
  const int wave = t >> 5;
  const int lane = t & 31;
  __shared__ alignas(16) bfloat sB[32 * 40];
  const int mrow = m0 + wave * 32;
  v8f acc00 = {0.f,0.f,0.f,0.f,0.f,0.f,0.f,0.f};
  v8f acc01 = acc00, acc10 = acc00, acc11 = acc00;
  for (int k0 = 0; k0 < Hk; k0 += 32) {
    if (t < 128) {
      int c  = t >> 2;        // 0..31 (output col within tile)
      int kq = (t & 3) * 8;   // 0,8,16,24
      *(v8bf*)(sB + c * 40 + kq) = *(const v8bf*)(W + (size_t)(cb0 + c) * Hk + k0 + kq);
    }
    __syncthreads();
    v16bf a0 = load_a_frag(A + (size_t)mrow * Hk + k0, Hk);
    v16bf a1 = load_a_frag(A + (size_t)(mrow + 16) * Hk + k0, Hk);
    v16bf b0 = load_b_frag(sB);
    v16bf b1 = load_b_frag(sB + 16 * 40);
    acc00 = wmma_bf16(a0, b0, acc00);
    acc01 = wmma_bf16(a0, b1, acc01);
    acc10 = wmma_bf16(a1, b0, acc10);
    acc11 = wmma_bf16(a1, b1, acc11);
    __syncthreads();
  }
  const int cc = lane & 15, hi = lane >> 4;
#pragma unroll
  for (int j = 0; j < 8; ++j) {
    int r0 = mrow + j + 8 * hi;
    int r1 = r0 + 16;
    int c0 = cb0 + cc, c1 = cb0 + 16 + cc;
    float v00 = acc00[j], v01 = acc01[j], v10 = acc10[j], v11 = acc11[j];
    if (bias) {
      v00 += bias[c0]; v01 += bias[c1]; v10 += bias[c0]; v11 += bias[c1];
    }
    if (addrow) {
      v00 += addrow[(size_t)r0 * GOk + c0]; v01 += addrow[(size_t)r0 * GOk + c1];
      v10 += addrow[(size_t)r1 * GOk + c0]; v11 += addrow[(size_t)r1 * GOk + c1];
    }
    out[(size_t)r0 * GOk + c0] = v00;
    out[(size_t)r0 * GOk + c1] = v01;
    out[(size_t)r1 * GOk + c0] = v10;
    out[(size_t)r1 * GOk + c1] = v11;
  }
}

__global__ __launch_bounds__(256) void k_lstm(const float* __restrict__ G,
                                              float* __restrict__ cx,
                                              float* __restrict__ hx,
                                              bfloat* __restrict__ hxb) {
  int idx = blockIdx.x * blockDim.x + threadIdx.x;  // < BN*H
  int r = idx >> 7, h = idx & (Hk - 1);
  const float* g = G + (size_t)r * GOk;
  float igv = sigmf(g[h]);
  float fgv = sigmf(g[Hk + h]);
  float ggv = tanhf(g[2 * Hk + h]);
  float ogv = sigmf(g[3 * Hk + h]);
  float c  = fgv * cx[idx] + igv * ggv;
  float hv = ogv * tanhf(c);
  cx[idx] = c;
  hx[idx] = hv;
  hxb[idx] = (bfloat)hv;
}

// ---------------------------------------------------------------- launch

extern "C" void kernel_launch(void* const* d_in, const int* in_sizes, int n_in,
                              void* d_out, int out_size, void* d_ws, size_t ws_size,
                              hipStream_t stream) {
  (void)in_sizes; (void)n_in; (void)out_size; (void)ws_size;
  const float* x    = (const float*)d_in[0];
  const float* adj  = (const float*)d_in[1];
  const float* W1   = (const float*)d_in[2];
  const float* b1   = (const float*)d_in[3];
  const float* W2   = (const float*)d_in[4];
  const float* b2   = (const float*)d_in[5];
  const float* W_ih = (const float*)d_in[6];
  const float* W_hh = (const float*)d_in[7];
  const float* b_ih = (const float*)d_in[8];
  const float* b_hh = (const float*)d_in[9];

  char* w = (char*)d_ws;
  auto carve = [&](size_t bytes) {
    char* p = w;
    w += (bytes + 255) & ~(size_t)255;
    return p;
  };
  bfloat* adjb    = (bfloat*)carve((size_t)Nk * Nk * 2);        // 2 MB
  bfloat* catb    = (bfloat*)carve((size_t)Bk * Nk * CPk * 2);  // 2.6 MB
  bfloat* acb     = (bfloat*)carve((size_t)Bk * Nk * CPk * 2);  // 2.6 MB
  bfloat* wcatb   = (bfloat*)carve((size_t)CPk * GOk * 2);
  bfloat* wihb    = (bfloat*)carve((size_t)GOk * Hk * 2);
  bfloat* whhb    = (bfloat*)carve((size_t)GOk * Hk * 2);
  float*  biascat = (float*)carve(GOk * 4);
  float*  biasdec = (float*)carve(GOk * 4);
  float*  G       = (float*)carve((size_t)BNk * GOk * 4);       // 16 MB
  float*  P       = (float*)carve((size_t)BNk * GOk * 4);       // 16 MB
  float*  hid     = (float*)carve((size_t)Bk * NHk * 4);        // 4 MB
  float*  ctx     = (float*)carve((size_t)Bk * NHk * 4);        // 4 MB
  bfloat* ctxb    = (bfloat*)carve((size_t)Bk * NHk * 2);
  float*  cx      = (float*)carve((size_t)BNk * Hk * 4);
  float*  hx      = (float*)carve((size_t)BNk * Hk * 4);
  bfloat* hxb     = (bfloat*)carve((size_t)BNk * Hk * 2);

  // one-time prep (runs every launch; deterministic)
  k_cvt_f2b<<<4096, 256, 0, stream>>>(adj, adjb, Nk * Nk);
  k_cvt_f2b<<<256, 256, 0, stream>>>(W_ih, wihb, GOk * Hk);
  k_cvt_f2b<<<256, 256, 0, stream>>>(W_hh, whhb, GOk * Hk);
  k_prep_wcat<<<(CPk * GOk) / 256, 256, 0, stream>>>(W1, W2, wcatb);
  k_prep_bias<<<2, 256, 0, stream>>>(b1, b2, b_ih, b_hh, biascat, biasdec);
  k_zero_f32<<<4096, 256, 0, stream>>>(hid, Bk * NHk);
  k_zero_f32<<<4096, 256, 0, stream>>>(ctx, Bk * NHk);
  k_zero_f32<<<4096, 256, 0, stream>>>(cx, BNk * Hk);
  k_zero_f32<<<4096, 256, 0, stream>>>(hx, BNk * Hk);
  k_zero_b16<<<4096, 256, 0, stream>>>(hxb, BNk * Hk);

  // encoder: 32 recurrent steps
  for (int tstep = 0; tstep < Sk; ++tstep) {
    k_build_cat<<<(Bk * Nk * CPk) / 256, 256, 0, stream>>>(x, hid, catb, tstep);
    k_gemm_adj<<<dim3(CPk / 32, Nk / 256, Bk), 256, 0, stream>>>(adjb, catb, acb);
    k_gemm_w<<<dim3(GOk / 32, BNk / 256), 256, 0, stream>>>(acb, wcatb, biascat, G);
    k_gates<<<dim3(NHk / 256, Bk), 256, 0, stream>>>(G, hid, ctx);
  }

  // decoder: ctx is constant across att steps -> precompute input projection once
  k_cvt_f2b<<<4096, 256, 0, stream>>>(ctx, ctxb, Bk * NHk);
  k_gemm_dec<<<dim3(GOk / 32, BNk / 256), 256, 0, stream>>>(ctxb, wihb, P, nullptr, biasdec);
  for (int it = 0; it < Sk; ++it) {
    k_gemm_dec<<<dim3(GOk / 32, BNk / 256), 256, 0, stream>>>(hxb, whhb, G, P, nullptr);
    k_lstm<<<4096, 256, 0, stream>>>(G, cx, hx, hxb);
  }

  hipMemcpyAsync(d_out, hx, (size_t)BNk * Hk * sizeof(float),
                 hipMemcpyDeviceToDevice, stream);
}
